// RelaxedWordMoverSimilarity_68917045232236
// MI455X (gfx1250) — matrix operand order repaired and checked by reading, not compiled, not run on hardware
//
#include <hip/hip_runtime.h>
#include <hip/hip_bf16.h>
#include <math.h>

// ---- types matching CDNA5 WMMA builtin signatures ----
typedef __attribute__((ext_vector_type(16))) __bf16         v16bf;
typedef __attribute__((ext_vector_type(8)))  __bf16         v8bf;
typedef __attribute__((ext_vector_type(16))) unsigned short v16u;
typedef __attribute__((ext_vector_type(8)))  float          v8f;
typedef __attribute__((ext_vector_type(8)))  float          f32x8;
typedef __attribute__((ext_vector_type(16))) float          f32x16;

#define EPSV   1e-8f
#define NEGINF (-__builtin_inff())

// RNE float -> bf16 bits
__device__ __forceinline__ unsigned short bf16_bits(float f) {
    unsigned u = __float_as_uint(f);
    u += 0x7FFFu + ((u >> 16) & 1u);
    return (unsigned short)(u >> 16);
}
// order-preserving float<->uint so integer atomicMax == float max
__device__ __forceinline__ unsigned ford(float f) {
    unsigned u = __float_as_uint(f);
    return (u & 0x80000000u) ? ~u : (u | 0x80000000u);
}
__device__ __forceinline__ float funord(unsigned u) {
    unsigned v = (u & 0x80000000u) ? (u & 0x7FFFFFFFu) : ~u;
    return __uint_as_float(v);
}

// ---------------- init: zero the ordered-max buffers ----------------
__global__ void init_kernel(unsigned* __restrict__ p, int n) {
    int i = blockIdx.x * blockDim.x + threadIdx.x;
    if (i < n) p[i] = 0u;   // 0 <= ford(-inf): identity for ordered max
}

// ------- fused fp32->bf16 convert + row norms: one wave per row -------
__global__ void convert_norms_kernel(const float* __restrict__ x,
                                     unsigned short* __restrict__ xb,
                                     float* __restrict__ n,
                                     int rows, int H, int do_convert) {
    int wave = threadIdx.x >> 5, lane = threadIdx.x & 31;
    int r = blockIdx.x * 8 + wave;
    if (r >= rows) return;
    const float* src = x + (size_t)r * H;
    float s = 0.0f;
    if (do_convert) {
        unsigned short* dst = xb + (size_t)r * H;
        for (int h = lane * 2; h < H; h += 64) {   // H assumed even
            float a = src[h], b = src[h + 1];
            s += a * a + b * b;
            unsigned pk = (unsigned)bf16_bits(a) | ((unsigned)bf16_bits(b) << 16);
            *(unsigned*)(dst + h) = pk;
        }
    } else {
        for (int h = lane; h < H; h += 32) { float v = src[h]; s += v * v; }
    }
#pragma unroll
    for (int m = 16; m >= 1; m >>= 1) s += __shfl_xor(s, m, 32);
    if (lane == 0) n[r] = sqrtf(s);
}

// ---------- shared epilogue: cosine normalize + masked row/col max ----------
__device__ __forceinline__ void epilogue(
    v8f acc[4], int b, int iw, int n0, int half, int col,
    const unsigned char* __restrict__ mask1, const unsigned char* __restrict__ mask2,
    const float* __restrict__ n1, const float* __restrict__ n2,
    unsigned* __restrict__ rowmax, unsigned* __restrict__ colmax,
    unsigned* ldscol, int S1, int S2)
{
    float n1v[8]; bool m1v[8];
#pragma unroll
    for (int r = 0; r < 8; ++r) {
        int row = iw + half * 8 + r;
        n1v[r] = n1[(size_t)b * S1 + row];
        m1v[r] = mask1[(size_t)b * S1 + row] != 0;
    }
    float rmax[8];
#pragma unroll
    for (int r = 0; r < 8; ++r) rmax[r] = NEGINF;

#pragma unroll
    for (int t = 0; t < 4; ++t) {
        int   j   = n0 + t * 16 + col;
        float n2v = n2[(size_t)b * S2 + j];
        bool  mj  = mask2[(size_t)b * S2 + j] != 0;
        float cmax = NEGINF;
#pragma unroll
        for (int r = 0; r < 8; ++r) {
            float denom = fmaxf(n1v[r] * n2v, EPSV);
            float s     = acc[t][r] / denom;
            s = (m1v[r] && mj) ? s : NEGINF;
            rmax[r] = fmaxf(rmax[r], s);
            cmax    = fmaxf(cmax, s);
        }
        cmax = fmaxf(cmax, __shfl_xor(cmax, 16, 32));       // combine M halves
        if (half == 0) atomicMax(&ldscol[t * 16 + col], ford(cmax));
    }

#pragma unroll
    for (int r = 0; r < 8; ++r) {                           // row max over 16 lanes in-half
        float v = rmax[r];
        v = fmaxf(v, __shfl_xor(v, 8, 32));
        v = fmaxf(v, __shfl_xor(v, 4, 32));
        v = fmaxf(v, __shfl_xor(v, 2, 32));
        v = fmaxf(v, __shfl_xor(v, 1, 32));
        if (col == 0)
            atomicMax(&rowmax[(size_t)b * S1 + iw + half * 8 + r], ford(v));
    }

    __syncthreads();
    if (threadIdx.x < 64)
        atomicMax(&colmax[(size_t)b * S2 + n0 + threadIdx.x], ldscol[threadIdx.x]);
}

// ======= PRIMARY: pre-converted bf16 GEMM — pure load_b128 -> v_wmma =======
// Block: 8 waves. Tile: 128 rows (16/wave) x 64 cols (4 WMMA tiles).
__global__ __launch_bounds__(256) void gemm_max_bf16_kernel(
    const unsigned short* __restrict__ x1b, const unsigned short* __restrict__ x2b,
    const unsigned char* __restrict__ mask1, const unsigned char* __restrict__ mask2,
    const float* __restrict__ n1, const float* __restrict__ n2,
    unsigned* __restrict__ rowmax, unsigned* __restrict__ colmax,
    int S1, int S2, int H)
{
    const int b    = blockIdx.z;
    const int i0   = blockIdx.y * 128;
    const int n0   = blockIdx.x * 64;
    const int wave = threadIdx.x >> 5;
    const int lane = threadIdx.x & 31;
    const int half = lane >> 4;
    const int col  = lane & 15;
    const int iw   = i0 + wave * 16;

    __shared__ unsigned ldscol[64];
    if (threadIdx.x < 64) ldscol[threadIdx.x] = 0u;
    __syncthreads();

    // A fragment (16x32 bf16): lane = row M=col; elems 0..15 = 16 contiguous K
    // at k0+half*16 -> two 16B loads (16B aligned)
    const unsigned short* Ap = x1b + ((size_t)b * S1 + iw + col) * H + half * 16;
    // B fragment (32x16 bf16): lane = col N=col; 16 contiguous K at k0+half*16
    // -> one 32B load (32B aligned: H*2 and half*16*2 are multiples of 32)
    const unsigned short* Bp[4];
#pragma unroll
    for (int t = 0; t < 4; ++t)
        Bp[t] = x2b + ((size_t)b * S2 + n0 + t * 16 + col) * H + half * 16;

    v8f acc[4];
#pragma unroll
    for (int t = 0; t < 4; ++t) acc[t] = (v8f)(0.0f);

    for (int k0 = 0; k0 < H; k0 += 32) {
        v8bf alo = *(const v8bf*)(Ap + k0);
        v8bf ahi = *(const v8bf*)(Ap + k0 + 8);
        v16bf a  = __builtin_shufflevector(alo, ahi,
                        0, 1, 2, 3, 4, 5, 6, 7, 8, 9, 10, 11, 12, 13, 14, 15);
#pragma unroll
        for (int t = 0; t < 4; ++t) {
            v16bf bb = *(const v16bf*)(Bp[t] + k0);
            acc[t] = __builtin_amdgcn_wmma_f32_16x16x32_bf16(
                false, a, false, bb, (short)0, acc[t], false, false);
        }
    }

    epilogue(acc, b, iw, n0, half, col, mask1, mask2, n1, n2,
             rowmax, colmax, ldscol, S1, S2);
}

// ======= FALLBACK (ws too small): convert-in-kernel fp32 GEMM =======
__global__ __launch_bounds__(256) void gemm_max_f32_kernel(
    const float* __restrict__ x1, const float* __restrict__ x2,
    const unsigned char* __restrict__ mask1, const unsigned char* __restrict__ mask2,
    const float* __restrict__ n1, const float* __restrict__ n2,
    unsigned* __restrict__ rowmax, unsigned* __restrict__ colmax,
    int S1, int S2, int H)
{
    const int b    = blockIdx.z;
    const int i0   = blockIdx.y * 128;
    const int n0   = blockIdx.x * 64;
    const int wave = threadIdx.x >> 5;
    const int lane = threadIdx.x & 31;
    const int half = lane >> 4;
    const int col  = lane & 15;
    const int iw   = i0 + wave * 16;

    __shared__ unsigned ldscol[64];
    if (threadIdx.x < 64) ldscol[threadIdx.x] = 0u;
    __syncthreads();

    const float* Ap = x1 + ((size_t)b * S1 + iw + col) * H + half * 16;
    const float* Bp[4];
#pragma unroll
    for (int t = 0; t < 4; ++t)
        Bp[t] = x2 + ((size_t)b * S2 + n0 + t * 16 + col) * H + half * 16;

    v8f acc[4];
#pragma unroll
    for (int t = 0; t < 4; ++t) acc[t] = (v8f)(0.0f);

    for (int k0 = 0; k0 < H; k0 += 32) {
        f32x8 alo = *(const f32x8*)(Ap + k0);
        f32x8 ahi = *(const f32x8*)(Ap + k0 + 8);
        v16u  au;
#pragma unroll
        for (int e = 0; e < 8; ++e) { au[e] = bf16_bits(alo[e]); au[8 + e] = bf16_bits(ahi[e]); }
        v16bf a = __builtin_bit_cast(v16bf, au);
#pragma unroll
        for (int t = 0; t < 4; ++t) {
            f32x16 bv = *(const f32x16*)(Bp[t] + k0);
            v16u   bu;
#pragma unroll
            for (int e = 0; e < 16; ++e) bu[e] = bf16_bits(bv[e]);
            v16bf bb = __builtin_bit_cast(v16bf, bu);
            acc[t] = __builtin_amdgcn_wmma_f32_16x16x32_bf16(
                false, a, false, bb, (short)0, acc[t], false, false);
        }
    }

    epilogue(acc, b, iw, n0, half, col, mask1, mask2, n1, n2,
             rowmax, colmax, ldscol, S1, S2);
}

// ---------------- finalize: masked means per batch ----------------
__global__ void finalize_kernel(const unsigned* __restrict__ rowmax, const unsigned* __restrict__ colmax,
                                const unsigned char* __restrict__ mask1, const unsigned char* __restrict__ mask2,
                                float* __restrict__ out, int S1, int S2) {
    int b = blockIdx.x, tid = threadIdx.x;
    __shared__ float rsum[256], rcnt[256];

    float sum = 0.0f, cnt = 0.0f;
    for (int i = tid; i < S1; i += 256)
        if (mask1[(size_t)b * S1 + i]) { sum += funord(rowmax[(size_t)b * S1 + i]); cnt += 1.0f; }
    rsum[tid] = sum; rcnt[tid] = cnt; __syncthreads();
    for (int s = 128; s > 0; s >>= 1) {
        if (tid < s) { rsum[tid] += rsum[tid + s]; rcnt[tid] += rcnt[tid + s]; }
        __syncthreads();
    }
    float m1 = rsum[0] / rcnt[0];
    __syncthreads();

    sum = 0.0f; cnt = 0.0f;
    for (int j = tid; j < S2; j += 256)
        if (mask2[(size_t)b * S2 + j]) { sum += funord(colmax[(size_t)b * S2 + j]); cnt += 1.0f; }
    rsum[tid] = sum; rcnt[tid] = cnt; __syncthreads();
    for (int s = 128; s > 0; s >>= 1) {
        if (tid < s) { rsum[tid] += rsum[tid + s]; rcnt[tid] += rcnt[tid + s]; }
        __syncthreads();
    }
    if (tid == 0) out[b] = 0.5f * (m1 + rsum[0] / rcnt[0]);
}

extern "C" void kernel_launch(void* const* d_in, const int* in_sizes, int n_in,
                              void* d_out, int out_size, void* d_ws, size_t ws_size,
                              hipStream_t stream) {
    const float*         x1    = (const float*)d_in[0];
    const unsigned char* mask1 = (const unsigned char*)d_in[1];
    const float*         x2    = (const float*)d_in[2];
    const unsigned char* mask2 = (const unsigned char*)d_in[3];
    float*               out   = (float*)d_out;

    const int B  = out_size;                                   // 32
    const int S1 = in_sizes[1] / B;                            // 512
    const int S2 = in_sizes[3] / B;                            // 512
    const int H  = (int)((long long)in_sizes[0] / ((long long)B * S1)); // 768

    const size_t R1 = (size_t)B * S1, R2 = (size_t)B * S2;

    // ws layout: n1 f32[R1] | n2 f32[R2] | rowmax u32[R1] | colmax u32[R2]
    //            | (256-aligned) x1b bf16[R1*H] | x2b bf16[R2*H]
    float*    n1     = (float*)d_ws;
    float*    n2     = n1 + R1;
    unsigned* rowmax = (unsigned*)(n2 + R2);
    unsigned* colmax = rowmax + R1;
    size_t small_bytes = (R1 + R2) * 2u * sizeof(unsigned);
    size_t pool_off    = (small_bytes + 255u) & ~(size_t)255u;
    size_t need        = pool_off + (R1 + R2) * (size_t)H * sizeof(unsigned short);
    const int use_pre  = (ws_size >= need);

    unsigned short* x1b = (unsigned short*)((char*)d_ws + pool_off);
    unsigned short* x2b = x1b + R1 * (size_t)H;

    const int totrc = (int)(R1 + R2);   // rowmax+colmax contiguous
    init_kernel<<<(totrc + 255) / 256, 256, 0, stream>>>(rowmax, totrc);

    convert_norms_kernel<<<((int)R1 + 7) / 8, 256, 0, stream>>>(x1, x1b, n1, (int)R1, H, use_pre);
    convert_norms_kernel<<<((int)R2 + 7) / 8, 256, 0, stream>>>(x2, x2b, n2, (int)R2, H, use_pre);

    dim3 grid(S2 / 64, S1 / 128, B);    // (8, 4, 32) = 1024 blocks, 8192 waves
    if (use_pre) {
        gemm_max_bf16_kernel<<<grid, 256, 0, stream>>>(x1b, x2b, mask1, mask2, n1, n2,
                                                       rowmax, colmax, S1, S2, H);
    } else {
        gemm_max_f32_kernel<<<grid, 256, 0, stream>>>(x1, x2, mask1, mask2, n1, n2,
                                                      rowmax, colmax, S1, S2, H);
    }

    finalize_kernel<<<B, 256, 0, stream>>>(rowmax, colmax, mask1, mask2, out, S1, S2);
}
